// WNO2d_23545010717197
// MI455X (gfx1250) — matrix operand order repaired and checked by reading, not compiled, not run on hardware
//
#include <hip/hip_runtime.h>
#include <hip/hip_fp16.h>

// ---------------------------------------------------------------------------
// WNO2d (wavelet neural operator) forward for MI455X / gfx1250.
// Channel-mixing contractions -> v_wmma_f32_16x16x32_f16 (f16 in, f32 acc).
// db4 analysis/synthesis filter banks -> coalesced VALU stencils.
// Whole working set (up to 192MiB per kernel) is L2-resident on the 192MB L2.
// ---------------------------------------------------------------------------

typedef __attribute__((ext_vector_type(16))) _Float16 v16h;
typedef __attribute__((ext_vector_type(8)))  float    v8f;

// db4 filters. Analysis kernels are the reversed decomposition filters
// (XLA conv is cross-correlation); synthesis kernels are the forward ones.
__constant__ float c_h0a[8] = { 0.23037781330885523f,  0.7148465705525415f,
                                0.6308807679295904f,  -0.02798376941698385f,
                               -0.18703481171888114f,  0.030841381835986965f,
                                0.032883011666982945f,-0.010597401784997278f };
__constant__ float c_h1a[8] = {-0.010597401784997278f,-0.032883011666982945f,
                                0.030841381835986965f, 0.18703481171888114f,
                               -0.02798376941698385f, -0.6308807679295904f,
                                0.7148465705525415f,  -0.23037781330885523f };
__constant__ float c_g0[8]  = {-0.010597401784997278f, 0.032883011666982945f,
                                0.030841381835986965f,-0.18703481171888114f,
                               -0.02798376941698385f,  0.6308807679295904f,
                                0.7148465705525415f,   0.23037781330885523f };
__constant__ float c_g1[8]  = {-0.23037781330885523f,  0.7148465705525415f,
                               -0.6308807679295904f,  -0.02798376941698385f,
                                0.18703481171888114f,  0.030841381835986965f,
                               -0.032883011666982945f,-0.010597401784997278f };

__device__ __forceinline__ float gelu_exact(float x) {
    return 0.5f * x * (1.0f + erff(x * 0.70710678118654752f));
}

// A fragment: rows m0..m0+15, K chunk k0..k0+31, from row-major W (ld = ldw).
// Layout (16-bit A 16x32): lane holds row (lane&15);
// lanes 0-15: e0..15 -> K 0..15; lanes 16-31: K 16..31 (matches B fill).
__device__ __forceinline__ v16h load_a_frag(const float* __restrict__ W,
                                            int m0, int k0, int ldw) {
    int lane = threadIdx.x & 31;
    int half = lane >> 4;
    int r    = lane & 15;
    const float* row = W + (size_t)(m0 + r) * ldw + k0;
    v16h a;
#pragma unroll
    for (int e = 0; e < 16; ++e) {
        int k = e + half * 16;
        a[e] = (_Float16)row[k];
    }
    return a;
}

// ---------------------------------------------------------------------------
// fc0 + grid channel: v[b,c,h,w] (NCHW, 256x256, zero-padded last row/col)
// ---------------------------------------------------------------------------
__global__ void fc0_kernel(const float* __restrict__ X,   // [4,255,255,1]
                           const float* __restrict__ W,   // [64,3]
                           const float* __restrict__ B,   // [64]
                           float* __restrict__ V)         // [4,64,256,256]
{
    size_t idx = (size_t)blockIdx.x * blockDim.x + threadIdx.x;
    const size_t total = (size_t)4 * 64 * 256 * 256;
    if (idx >= total) return;
    int w = (int)(idx & 255);
    int h = (int)(idx >> 8) & 255;
    int c = (int)(idx >> 16) & 63;
    int b = (int)(idx >> 22);
    float v = 0.0f;
    if (h < 255 && w < 255) {
        float x  = X[((size_t)b * 255 + h) * 255 + w];
        float gx = (float)h * (1.0f / 254.0f);
        float gy = (float)w * (1.0f / 254.0f);
        v = W[c * 3 + 0] * x + W[c * 3 + 1] * gx + W[c * 3 + 2] * gy + B[c];
    }
    V[idx] = v;
}

// ---------------------------------------------------------------------------
// DWT analysis, row direction (axis=W). Symmetric ext over Nz samples
// (Nz = Win or Win+1 with implicit trailing zero), padL == 6 for all levels.
// ---------------------------------------------------------------------------
__global__ void dwt_row(const float* __restrict__ X, float* __restrict__ Lo,
                        float* __restrict__ Hi, int rows, int Win, int Wout,
                        int Nz)
{
    int idx = blockIdx.x * blockDim.x + threadIdx.x;
    int total = rows * Wout;
    if (idx >= total) return;
    int t = idx % Wout;
    int r = idx / Wout;
    const float* x = X + (size_t)r * Win;
    float lo = 0.0f, hi = 0.0f;
#pragma unroll
    for (int j = 0; j < 8; ++j) {
        int q = 2 * t + j - 6;
        if (q < 0) q = -q - 1;
        else if (q >= Nz) q = 2 * Nz - q - 1;
        float xv = (q < Win) ? x[q] : 0.0f;   // trailing zero-pad sample
        lo += c_h0a[j] * xv;
        hi += c_h1a[j] * xv;
    }
    Lo[idx] = lo;
    Hi[idx] = hi;
}

// DWT analysis, column direction (axis=H). Input [C2,Hin,W] contiguous.
__global__ void dwt_col(const float* __restrict__ X, float* __restrict__ Lo,
                        float* __restrict__ Hi, int C2, int Hin, int Hout,
                        int W, int Nz)
{
    int idx = blockIdx.x * blockDim.x + threadIdx.x;
    int total = C2 * Hout * W;
    if (idx >= total) return;
    int w  = idx % W;
    int t  = (idx / W) % Hout;
    int c2 = idx / (W * Hout);
    const float* xp = X + (size_t)c2 * Hin * W + w;
    float lo = 0.0f, hi = 0.0f;
#pragma unroll
    for (int j = 0; j < 8; ++j) {
        int q = 2 * t + j - 6;
        if (q < 0) q = -q - 1;
        else if (q >= Nz) q = 2 * Nz - q - 1;
        float xv = (q < Hin) ? xp[(size_t)q * W] : 0.0f;
        lo += c_h0a[j] * xv;
        hi += c_h1a[j] * xv;
    }
    Lo[idx] = lo;
    Hi[idx] = hi;
}

// ---------------------------------------------------------------------------
// IDWT synthesis, column direction: out[t] = sum over even q=t+j-1 of
// g0[j]*lo[q/2] + g1[j]*hi[q/2]  (lhs_dilation=2, pad=1, db4 transpose conv).
// Lo/Hi may be larger buffers (crop via plane/row strides). Out [C2,2N-6,W].
// ---------------------------------------------------------------------------
__global__ void idwt_col(const float* __restrict__ Lo, int loPlane, int loRow,
                         const float* __restrict__ Hi, int hiPlane, int hiRow,
                         float* __restrict__ Out, int C2, int Nin, int W)
{
    int Hout = 2 * Nin - 6;
    int idx = blockIdx.x * blockDim.x + threadIdx.x;
    int total = C2 * Hout * W;
    if (idx >= total) return;
    int w  = idx % W;
    int t  = (idx / W) % Hout;
    int c2 = idx / (W * Hout);
    float acc = 0.0f;
#pragma unroll
    for (int j = 0; j < 8; ++j) {
        int q = t + j - 1;
        if (q >= 0 && q <= 2 * Nin - 2 && (q & 1) == 0) {
            int m = q >> 1;
            acc += c_g0[j] * Lo[(size_t)c2 * loPlane + (size_t)m * loRow + w]
                 + c_g1[j] * Hi[(size_t)c2 * hiPlane + (size_t)m * hiRow + w];
        }
    }
    Out[idx] = acc;
}

// IDWT synthesis, row direction. Lo/Hi contiguous [rows, Nin]; Out [rows, 2N-6].
__global__ void idwt_row(const float* __restrict__ Lo,
                         const float* __restrict__ Hi,
                         float* __restrict__ Out, int rows, int Nin)
{
    int Wout = 2 * Nin - 6;
    int idx = blockIdx.x * blockDim.x + threadIdx.x;
    int total = rows * Wout;
    if (idx >= total) return;
    int t = idx % Wout;
    int r = idx / Wout;
    const float* lp = Lo + (size_t)r * Nin;
    const float* hp = Hi + (size_t)r * Nin;
    float acc = 0.0f;
#pragma unroll
    for (int j = 0; j < 8; ++j) {
        int q = t + j - 1;
        if (q >= 0 && q <= 2 * Nin - 2 && (q & 1) == 0) {
            int m = q >> 1;
            acc += c_g0[j] * lp[m] + c_g1[j] * hp[m];
        }
    }
    Out[idx] = acc;
}

// ---------------------------------------------------------------------------
// Skip GEMM: Y[b,o,hw] = maybe_gelu( sum_i Wk[o,i]*X[b,i,hw] + bk[o] + S[b,o,hw] )
// One wave per 64(M) x 16(N) tile; K=64 -> 2 WMMAs per M-tile, 8 per wave.
// ---------------------------------------------------------------------------
__global__ void __launch_bounds__(32)
skip_gemm64(const float* __restrict__ X, const float* __restrict__ Wk,
            const float* __restrict__ bk, const float* __restrict__ S,
            float* __restrict__ Y, int do_gelu)
{
    const int HW = 65536;
    int lane = threadIdx.x & 31;
    int half = lane >> 4;
    int c16  = lane & 15;
    int n0   = blockIdx.x * 16;
    int b    = blockIdx.y;
    const float* Xb = X + (size_t)b * 64 * HW;
    const float* Sb = S + (size_t)b * 64 * HW;
    float*       Yb = Y + (size_t)b * 64 * HW;

    v16h Bf[2];
#pragma unroll
    for (int c = 0; c < 2; ++c) {
        int kbase = c * 32 + half * 16;
#pragma unroll
        for (int e = 0; e < 16; ++e)
            Bf[c][e] = (_Float16)Xb[(size_t)(kbase + e) * HW + n0 + c16];
    }
#pragma unroll
    for (int mt = 0; mt < 4; ++mt) {
        v8f acc = {};
#pragma unroll
        for (int c = 0; c < 2; ++c) {
            v16h Af = load_a_frag(Wk, mt * 16, c * 32, 64);
            acc = __builtin_amdgcn_wmma_f32_16x16x32_f16(
                false, Af, false, Bf[c], (short)0, acc, false, false);
        }
#pragma unroll
        for (int r = 0; r < 8; ++r) {
            int row = mt * 16 + r + half * 8;
            int n   = n0 + c16;
            float v = acc[r] + bk[row] + Sb[(size_t)row * HW + n];
            if (do_gelu) v = gelu_exact(v);
            Yb[(size_t)row * HW + n] = v;
        }
    }
}

// ---------------------------------------------------------------------------
// Spectral mode-mixing: out[b,o,xy] = sum_i in[b,i,xy] * W[i,o,xy]
// One wave per (x,y) mode; M=64, K=64, N=16 (batch 4 in cols 0..3).
// ---------------------------------------------------------------------------
__global__ void __launch_bounds__(32)
spectral_gemm64(const float* __restrict__ IN, const float* __restrict__ W,
                float* __restrict__ OUT)
{
    const int SS = 484;               // 22*22 modes
    int xy   = blockIdx.x;
    int lane = threadIdx.x & 31;
    int half = lane >> 4;
    int c16  = lane & 15;

    v16h Bf[2];
#pragma unroll
    for (int c = 0; c < 2; ++c) {
        int kbase = c * 32 + half * 16;
#pragma unroll
        for (int e = 0; e < 16; ++e) {
            float x = (c16 < 4)
                ? IN[((size_t)(c16 * 64 + kbase + e)) * SS + xy] : 0.0f;
            Bf[c][e] = (_Float16)x;
        }
    }
    int r15 = lane & 15;
#pragma unroll
    for (int mt = 0; mt < 4; ++mt) {
        v8f acc = {};
#pragma unroll
        for (int c = 0; c < 2; ++c) {
            v16h Af;
#pragma unroll
            for (int e = 0; e < 16; ++e) {
                int k = e + half * 16 + c * 32;
                Af[e] = (_Float16)W[((size_t)(k * 64 + mt * 16 + r15)) * SS + xy];
            }
            acc = __builtin_amdgcn_wmma_f32_16x16x32_f16(
                false, Af, false, Bf[c], (short)0, acc, false, false);
        }
        if (c16 < 4) {
#pragma unroll
            for (int r = 0; r < 8; ++r) {
                int row = mt * 16 + r + half * 8;
                OUT[((size_t)(c16 * 64 + row)) * SS + xy] = acc[r];
            }
        }
    }
}

// ---------------------------------------------------------------------------
// Head: out[b,h,w] += sum_o fc2_w[o]*gelu(sum_c fc1_w[o,c]*V[b,c,hw] + fc1_b[o]) + fc2_b
// M=192 -> 12 M-tiles, fc2 reduction done in-register + cross-half shuffle.
// ---------------------------------------------------------------------------
__global__ void __launch_bounds__(32)
head_fc1_fc2(const float* __restrict__ V, const float* __restrict__ W1,
             const float* __restrict__ b1, const float* __restrict__ W2,
             const float* __restrict__ b2, float* __restrict__ out,
             int accumulate)
{
    const int HW = 65536;
    int lane = threadIdx.x & 31;
    int half = lane >> 4;
    int c16  = lane & 15;
    int n0   = blockIdx.x * 16;
    int b    = blockIdx.y;
    const float* Vb = V + (size_t)b * 64 * HW;

    v16h Bf[2];
#pragma unroll
    for (int c = 0; c < 2; ++c) {
        int kbase = c * 32 + half * 16;
#pragma unroll
        for (int e = 0; e < 16; ++e)
            Bf[c][e] = (_Float16)Vb[(size_t)(kbase + e) * HW + n0 + c16];
    }
    float partial = 0.0f;
#pragma unroll
    for (int mt = 0; mt < 12; ++mt) {
        v8f acc = {};
#pragma unroll
        for (int c = 0; c < 2; ++c) {
            v16h Af = load_a_frag(W1, mt * 16, c * 32, 64);
            acc = __builtin_amdgcn_wmma_f32_16x16x32_f16(
                false, Af, false, Bf[c], (short)0, acc, false, false);
        }
#pragma unroll
        for (int r = 0; r < 8; ++r) {
            int row = mt * 16 + r + half * 8;
            float u = gelu_exact(acc[r] + b1[row]);
            partial += u * W2[row];
        }
    }
    partial += __shfl_xor(partial, 16, 32);   // combine the two row-halves
    if (half == 0) {
        int n = n0 + c16;
        int h = n >> 8, w = n & 255;
        if (h < 255 && w < 255) {
            size_t idx = ((size_t)b * 255 + h) * 255 + w;
            float v = partial + b2[0];
            if (accumulate) out[idx] += v;
            else            out[idx]  = v;
        }
    }
}

// ---------------------------------------------------------------------------
// Host-side orchestration
// ---------------------------------------------------------------------------
extern "C" void kernel_launch(void* const* d_in, const int* in_sizes, int n_in,
                              void* d_out, int out_size, void* d_ws,
                              size_t ws_size, hipStream_t stream)
{
    (void)in_sizes; (void)n_in; (void)out_size; (void)ws_size;
    const float* X = (const float*)d_in[0];
    float* out = (float*)d_out;
    float* ws  = (float*)d_ws;

    const int OS[5] = {256, 131, 69, 38, 22};  // level sizes
    const int NZ[4] = {256, 132, 70, 38};      // zero-extended lengths
    const size_t C2 = 256;                     // B*WIDTH = 4*64
    const size_t HW = 65536;                   // 256*256

    size_t off = 0;
    auto alloc = [&](size_t n) { size_t o = off; off += n; return o; };
    size_t o_va = alloc(C2 * HW);
    size_t o_vb = alloc(C2 * HW);
    size_t o_s  = alloc(C2 * HW);
    size_t o_ta = alloc(C2 * 256 * 131);
    size_t o_tb = alloc(C2 * 256 * 131);
    size_t o_ll[4], o_lh[4], o_hl[4], o_hh[4];
    for (int l = 0; l < 4; ++l) {
        size_t sz = C2 * (size_t)OS[l + 1] * OS[l + 1];
        o_ll[l] = alloc(sz); o_lh[l] = alloc(sz);
        o_hl[l] = alloc(sz); o_hh[l] = alloc(sz);
    }
    size_t o_outll  = alloc(C2 * 484);
    size_t o_band0  = alloc(C2 * 484);
    size_t o_rec38  = alloc(C2 * 38 * 38);
    size_t o_rec70  = alloc(C2 * 70 * 70);
    size_t o_rec132 = alloc(C2 * 132 * 132);

    float* vA = ws + o_va;
    float* vB = ws + o_vb;
    float* sB = ws + o_s;
    float* tA = ws + o_ta;
    float* tB = ws + o_tb;

    for (int br = 0; br < 2; ++br) {
        const float* const* P = (const float* const*)(d_in + 1 + br * 22);
        const float* fc0_w = P[0];  const float* fc0_b = P[1];
        const float* fc1_w = P[2];  const float* fc1_b = P[3];
        const float* fc2_w = P[4];  const float* fc2_b = P[5];

        {   // lift + grid
            size_t total = C2 * HW;
            fc0_kernel<<<(unsigned)((total + 255) / 256), 256, 0, stream>>>(
                X, fc0_w, fc0_b, vA);
        }

        for (int layer = 0; layer < 4; ++layer) {
            const float* w1 = P[6 + 4 * layer];
            const float* w2 = P[7 + 4 * layer];
            const float* kw = P[8 + 4 * layer];
            const float* kb = P[9 + 4 * layer];

            // ---- DWT (J=4) ----
            const float* cur = vA;
            for (int l = 0; l < 4; ++l) {
                int inS = OS[l], outS = OS[l + 1], nz = NZ[l];
                int rows = (int)C2 * inS;
                int totR = rows * outS;
                dwt_row<<<(totR + 255) / 256, 256, 0, stream>>>(
                    cur, tA, tB, rows, inS, outS, nz);
                int totC = (int)C2 * outS * outS;
                dwt_col<<<(totC + 255) / 256, 256, 0, stream>>>(
                    tA, ws + o_ll[l], ws + o_lh[l], (int)C2, inS, outS, outS, nz);
                dwt_col<<<(totC + 255) / 256, 256, 0, stream>>>(
                    tB, ws + o_hl[l], ws + o_hh[l], (int)C2, inS, outS, outS, nz);
                cur = ws + o_ll[l];
            }

            // ---- spectral 64x64 mode mixing (WMMA) ----
            spectral_gemm64<<<484, 32, 0, stream>>>(ws + o_ll[3], w1, ws + o_outll);
            spectral_gemm64<<<484, 32, 0, stream>>>(ws + o_lh[3], w2, ws + o_band0);

            // ---- IDWT ----
            const float* recP = nullptr;
            int recW = 0;
            for (int l = 3; l >= 0; --l) {
                int S  = OS[l + 1];
                int Hc = 2 * S - 6;
                const float* llp; int llPlane, llRow; const float* lhp;
                if (l == 3) { llp = ws + o_outll; llPlane = 484; llRow = 22;
                              lhp = ws + o_band0; }
                else        { llp = recP; llPlane = recW * recW; llRow = recW;
                              lhp = ws + o_lh[l]; }
                int bandPlane = S * S;
                int totC = (int)C2 * Hc * S;
                idwt_col<<<(totC + 255) / 256, 256, 0, stream>>>(
                    llp, llPlane, llRow, lhp, bandPlane, S, tA, (int)C2, S, S);
                idwt_col<<<(totC + 255) / 256, 256, 0, stream>>>(
                    ws + o_hl[l], bandPlane, S, ws + o_hh[l], bandPlane, S,
                    tB, (int)C2, S, S);
                float* outP;
                if (l == 0)      outP = sB;
                else if (l == 3) outP = ws + o_rec38;
                else if (l == 2) outP = ws + o_rec70;
                else             outP = ws + o_rec132;
                int rowsR = (int)C2 * Hc;
                int totRo = rowsR * (2 * S - 6);
                idwt_row<<<(totRo + 255) / 256, 256, 0, stream>>>(
                    tA, tB, outP, rowsR, S);
                recP = outP;
                recW = 2 * S - 6;
            }

            // ---- skip GEMM + residual add + gelu (WMMA) ----
            dim3 g((unsigned)(HW / 16), 4);
            skip_gemm64<<<g, 32, 0, stream>>>(vA, kw, kb, sB, vB,
                                              (layer < 3) ? 1 : 0);
            float* t = vA; vA = vB; vB = t;
        }

        // ---- fused fc1 + gelu + fc2 head (WMMA) ----
        dim3 g((unsigned)(HW / 16), 4);
        head_fc1_fc2<<<g, 32, 0, stream>>>(vA, fc1_w, fc1_b, fc2_w, fc2_b,
                                           out, br);
    }
}
